// RegionProposalLayer_9251359556278
// MI455X (gfx1250) — matrix operand order repaired and checked by reading, not compiled, not run on hardware
//
#include <hip/hip_runtime.h>
#include <hip/hip_bf16.h>

typedef __attribute__((ext_vector_type(16))) __bf16 v16bf;
typedef __attribute__((ext_vector_type(8)))  __bf16 v8bf;
typedef __attribute__((ext_vector_type(8)))  float  v8f;

#define BATCH 16
#define CH    512
#define HH    50
#define WW    76
#define HWSZ  3800            // 50*76
#define NPIX  60800           // 16*3800
#define KCONV 4608            // 512*9
#define HP    52              // H+2
#define WP    78              // W+2
#define OUT1  31129600UL      // offset of cls scores  (16*512*3800)
#define OUT2  32224000UL      // offset of bbox deltas (OUT1 + 16*18*3800)

__device__ __forceinline__ unsigned pack_bf16x2(float a, float b) {
  unsigned short ua = __builtin_bit_cast(unsigned short, (__bf16)a);
  unsigned short ub = __builtin_bit_cast(unsigned short, (__bf16)b);
  return (unsigned)ua | ((unsigned)ub << 16);
}

// ---------------------------------------------------------------------------
// Prep 1: W_conv (O,I,3,3) fp32 -> Wt[o][tap*512 + c] bf16  (k contiguous)
// ---------------------------------------------------------------------------
__global__ void prep_wconv(const float* __restrict__ w, __bf16* __restrict__ wt) {
  int idx = blockIdx.x * 256 + threadIdx.x;          // 512*4608 exact
  int o = idx / KCONV;
  int k = idx - o * KCONV;
  int tap = k >> 9, c = k & 511;
  int ky = (tap * 11) >> 5;                           // tap/3 for tap in 0..8
  int kx = tap - ky * 3;
  wt[idx] = (__bf16)w[((size_t)(o * CH + c) * 3 + ky) * 3 + kx];
}

// ---------------------------------------------------------------------------
// Prep 2: pack W_cls (18,512) + W_bbox (36,512) into Whead[64][512] bf16, rows
// 54..63 zero-padded so the head GEMM can run full 64-wide WMMA tiles.
// ---------------------------------------------------------------------------
__global__ void prep_whead(const float* __restrict__ wcls,
                           const float* __restrict__ wbbox,
                           __bf16* __restrict__ wh) {
  int idx = blockIdx.x * 256 + threadIdx.x;          // 64*512 exact
  int r = idx >> 9, c = idx & 511;
  float v = 0.f;
  if (r < 18)      v = wcls[r * CH + c];
  else if (r < 54) v = wbbox[(r - 18) * CH + c];
  wh[idx] = (__bf16)v;
}

// ---------------------------------------------------------------------------
// Prep 3: features NCHW fp32 -> zero-padded NHWC bf16 [B][52][78][512]
// (borders pre-zeroed by hipMemsetAsync). 32x32 LDS tile transpose so both
// the NCHW reads and the NHWC writes are coalesced.
// ---------------------------------------------------------------------------
__global__ void nchw_to_padded_nhwc(const float* __restrict__ in,
                                    __bf16* __restrict__ out) {
  __shared__ float tile[32][33];
  const int b  = blockIdx.z;
  const int c0 = blockIdx.y * 32;
  const int s0 = blockIdx.x * 32;
  const int tx = threadIdx.x, ty = threadIdx.y;
  #pragma unroll
  for (int k = 0; k < 32; k += 8) {
    int s = s0 + tx;
    float v = 0.f;
    if (s < HWSZ) v = in[((size_t)(b * CH + c0 + ty + k)) * HWSZ + s];
    tile[ty + k][tx] = v;
  }
  __syncthreads();
  #pragma unroll
  for (int k = 0; k < 32; k += 8) {
    int s = s0 + ty + k;
    if (s < HWSZ) {
      int y = s / WW, x = s - y * WW;
      size_t o = (((size_t)b * HP + (y + 1)) * WP + (x + 1)) * CH + c0 + tx;
      out[o] = (__bf16)tile[ty + k][tx];
    }
  }
}

// ---------------------------------------------------------------------------
// Main 3x3 conv as implicit GEMM:  M=512 out-ch, N=60800 pixels, K=4608.
// Block = 4 waves (2x2), wave tile 64x64, K step 32 via v_wmma_f32_16x16x32_bf16.
// A (weights) per lane: row o=L%16, K chunks at (L/16)*16 and +8 (two b128s).
// B (features) per lane: pixel n=L%16, 16 contiguous channels at (L/16)*16.
// Epilogue: ReLU -> coalesced fp32 NCHW store + LDS-transposed bf16 NHWC store.
// ---------------------------------------------------------------------------
__launch_bounds__(128)
__global__ void conv3x3_gemm(const __bf16* __restrict__ featPad,
                             const __bf16* __restrict__ Wt,
                             float* __restrict__ outF,
                             __bf16* __restrict__ featOut) {
  __shared__ __align__(16) unsigned ldsT[4][64][36];  // [wave][px][32 uints + pad]

  const int lane  = threadIdx.x & 31;
  const int wave  = threadIdx.x >> 5;
  const int wm    = wave >> 1, wn = wave & 1;
  const int lmod  = lane & 15, lhalf = lane >> 4;
  const int o_base = blockIdx.y * 128 + wm * 64;      // 512/128 = 4 exact
  const int p_base = blockIdx.x * 128 + wn * 64;      // 60800/128 = 475 exact

  const __bf16* aPtr[4];
  #pragma unroll
  for (int i = 0; i < 4; ++i)
    aPtr[i] = Wt + (size_t)(o_base + 16 * i + lmod) * KCONV + lhalf * 16;

  const __bf16* bPtr[4];
  int pixB[4], pixS[4];
  #pragma unroll
  for (int j = 0; j < 4; ++j) {
    int p = p_base + 16 * j + lmod;
    int b = p / HWSZ, s = p - b * HWSZ;
    int y = s / WW,   x = s - y * WW;
    pixB[j] = b; pixS[j] = s;
    // base at padded[y][x]; taps (ky,kx) in 0..2 reach padded[y+2][x+2]
    bPtr[j] = featPad + (((size_t)b * HP + y) * WP + x) * CH + lhalf * 16;
  }

  v8f acc[4][4] = {};

  for (int k0 = 0; k0 < KCONV; k0 += 32) {
    int tap = k0 >> 9;
    int ky  = (tap * 11) >> 5;
    int kx  = tap - ky * 3;
    int off = (ky * WP + kx) * CH + (k0 & 511);

    v16bf bb[4];
    #pragma unroll
    for (int j = 0; j < 4; ++j)
      bb[j] = *(const v16bf*)(bPtr[j] + off);

    v16bf aa[4];
    #pragma unroll
    for (int i = 0; i < 4; ++i) {
      v8bf lo = *(const v8bf*)(aPtr[i] + k0);
      v8bf hi = *(const v8bf*)(aPtr[i] + k0 + 8);
      v16bf t;
      #pragma unroll
      for (int e = 0; e < 8; ++e) { t[e] = lo[e]; t[e + 8] = hi[e]; }
      aa[i] = t;
    }

    #pragma unroll
    for (int i = 0; i < 4; ++i)
      #pragma unroll
      for (int j = 0; j < 4; ++j)
        acc[i][j] = __builtin_amdgcn_wmma_f32_16x16x32_bf16(
            false, aa[i], false, bb[j], (short)0, acc[i][j], false, false);
  }

  // Epilogue: ReLU, fp32 NCHW (coalesced: lanes vary pixel -> stride 1),
  // bf16 into LDS for coalesced NHWC write-out.
  #pragma unroll
  for (int j = 0; j < 4; ++j) {
    size_t ob = ((size_t)pixB[j] * CH) * HWSZ + pixS[j];
    int lp = 16 * j + lmod;
    #pragma unroll
    for (int i = 0; i < 4; ++i) {
      int o0 = o_base + 16 * i + 8 * lhalf;
      float v[8];
      #pragma unroll
      for (int r = 0; r < 8; ++r) v[r] = fmaxf(acc[i][j][r], 0.0f);
      #pragma unroll
      for (int r = 0; r < 8; ++r) outF[ob + (size_t)(o0 + r) * HWSZ] = v[r];
      unsigned* dst = &ldsT[wave][lp][8 * i + 4 * lhalf];
      dst[0] = pack_bf16x2(v[0], v[1]);
      dst[1] = pack_bf16x2(v[2], v[3]);
      dst[2] = pack_bf16x2(v[4], v[5]);
      dst[3] = pack_bf16x2(v[6], v[7]);
    }
  }
  __syncthreads();
  // NHWC bf16 store: per pass each lane moves 16B; 8 lanes cover one pixel's
  // 64 channels = one aligned 128B line.
  #pragma unroll
  for (int q = 0; q < 16; ++q) {
    int u  = q * 128 + lane * 4;
    int lp = u >> 5;
    int cu = u & 31;
    const unsigned* src = &ldsT[wave][lp][cu];
    uint4 d;
    d.x = src[0]; d.y = src[1]; d.z = src[2]; d.w = src[3];
    size_t p = (size_t)(p_base + lp);
    *(uint4*)((__bf16*)featOut + p * CH + o_base + cu * 2) = d;
  }
}

// ---------------------------------------------------------------------------
// 1x1 heads as one GEMM: M=64 (18 cls + 36 bbox + 10 zero pad), N=60800, K=512.
// One wave per 64-pixel tile; bias + scatter to the two NCHW outputs.
// ---------------------------------------------------------------------------
__launch_bounds__(64)
__global__ void heads_gemm(const __bf16* __restrict__ featOut,
                           const __bf16* __restrict__ Wh,
                           const float* __restrict__ bcls,
                           const float* __restrict__ bbbox,
                           float* __restrict__ out) {
  const int lane  = threadIdx.x & 31;
  const int wave  = threadIdx.x >> 5;
  const int lmod  = lane & 15, lhalf = lane >> 4;
  const int p_base = blockIdx.x * 128 + wave * 64;    // 60800/128 = 475 exact

  const __bf16* aPtr[4];
  #pragma unroll
  for (int i = 0; i < 4; ++i)
    aPtr[i] = Wh + (16 * i + lmod) * CH + lhalf * 16;

  const __bf16* bPtr[4];
  #pragma unroll
  for (int j = 0; j < 4; ++j)
    bPtr[j] = featOut + (size_t)(p_base + 16 * j + lmod) * CH + lhalf * 16;

  v8f acc[4][4] = {};

  for (int k0 = 0; k0 < CH; k0 += 32) {
    v16bf bb[4];
    #pragma unroll
    for (int j = 0; j < 4; ++j)
      bb[j] = *(const v16bf*)(bPtr[j] + k0);
    v16bf aa[4];
    #pragma unroll
    for (int i = 0; i < 4; ++i) {
      v8bf lo = *(const v8bf*)(aPtr[i] + k0);
      v8bf hi = *(const v8bf*)(aPtr[i] + k0 + 8);
      v16bf t;
      #pragma unroll
      for (int e = 0; e < 8; ++e) { t[e] = lo[e]; t[e + 8] = hi[e]; }
      aa[i] = t;
    }
    #pragma unroll
    for (int i = 0; i < 4; ++i)
      #pragma unroll
      for (int j = 0; j < 4; ++j)
        acc[i][j] = __builtin_amdgcn_wmma_f32_16x16x32_bf16(
            false, aa[i], false, bb[j], (short)0, acc[i][j], false, false);
  }

  #pragma unroll
  for (int j = 0; j < 4; ++j) {
    int p = p_base + 16 * j + lmod;
    int b = p / HWSZ, s = p - b * HWSZ;
    #pragma unroll
    for (int i = 0; i < 4; ++i) {
      int o0 = 16 * i + 8 * lhalf;
      #pragma unroll
      for (int r = 0; r < 8; ++r) {
        int o = o0 + r;
        if (o >= 54) continue;                        // zero-pad rows
        if (o < 18) {
          float v = acc[i][j][r] + bcls[o];
          out[OUT1 + ((size_t)(b * 18 + o)) * HWSZ + s] = v;
        } else {
          float v = acc[i][j][r] + bbbox[o - 18];
          out[OUT2 + ((size_t)(b * 36 + (o - 18))) * HWSZ + s] = v;
        }
      }
    }
  }
}

// ---------------------------------------------------------------------------
extern "C" void kernel_launch(void* const* d_in, const int* in_sizes, int n_in,
                              void* d_out, int out_size, void* d_ws, size_t ws_size,
                              hipStream_t stream) {
  const float* features = (const float*)d_in[0];
  const float* Wconv    = (const float*)d_in[1];
  const float* Wcls     = (const float*)d_in[2];
  const float* bcls     = (const float*)d_in[3];
  const float* Wbbox    = (const float*)d_in[4];
  const float* bbbox    = (const float*)d_in[5];
  float* out = (float*)d_out;
  char*  ws  = (char*)d_ws;

  // Workspace layout (all 256B aligned), total ~127.3 MB:
  const size_t FEATPAD_BYTES = (size_t)BATCH * HP * WP * CH * 2;   // 66,453,504
  __bf16* featPad = (__bf16*)(ws);
  __bf16* Wt      = (__bf16*)(ws + 66453504UL);                     // 4,718,592
  __bf16* Wh      = (__bf16*)(ws + 71172096UL);                     //    65,536
  __bf16* featOut = (__bf16*)(ws + 71237632UL);                     // 62,259,200

  hipMemsetAsync(featPad, 0, FEATPAD_BYTES, stream);                // zero borders
  prep_wconv<<<9216, 256, 0, stream>>>(Wconv, Wt);                  // 512*4608/256
  prep_whead<<<128, 256, 0, stream>>>(Wcls, Wbbox, Wh);             // 64*512/256
  nchw_to_padded_nhwc<<<dim3(119, 16, 16), dim3(32, 8), 0, stream>>>(features, featPad);
  conv3x3_gemm<<<dim3(475, 4), 128, 0, stream>>>(featPad, Wt, out, featOut);
  heads_gemm<<<475, 64, 0, stream>>>(featOut, Wh, bcls, bbbox, out);
}
